// CrossBlock_35794257445308
// MI455X (gfx1250) — compile-verified, matrix operand order checked by reading, not run on hardware
//
#include <hip/hip_runtime.h>

typedef __attribute__((ext_vector_type(16))) _Float16 v16h;
typedef __attribute__((ext_vector_type(8)))  _Float16 v8h;
typedef __attribute__((ext_vector_type(8)))  float    v8f;
typedef unsigned int u32;
typedef unsigned long long u64;
typedef __attribute__((ext_vector_type(4))) u32 u32x4;
typedef __attribute__((ext_vector_type(8))) u32 u32x8;

#define WMMA_F32_F16(a, b, c) \
  __builtin_amdgcn_wmma_f32_16x16x32_f16(false, (a), false, (b), (short)0, (c), false, false)

__device__ __forceinline__ v8f zero8() {
  v8f z = {0.f, 0.f, 0.f, 0.f, 0.f, 0.f, 0.f, 0.f};
  return z;
}

// Load one 16x32 f16 A/B fragment from an LDS row (row-major, contiguous K).
// Per the CDNA5 VGPR layout: lanes 0-15 hold K={0..7,16..23}, lanes 16-31 K={8..15,24..31}.
__device__ __forceinline__ v16h lds_frag(const _Float16* rowp, int kbase, int hf) {
  v8h lo = *(const v8h*)(rowp + kbase + hf * 8);
  v8h hi = *(const v8h*)(rowp + kbase + 16 + hf * 8);
  v16h r;
#pragma unroll
  for (int i = 0; i < 8; i++) { r[i] = lo[i]; r[i + 8] = hi[i]; }
  return r;
}

// ---------------------------------------------------------------------------
// Generic WMMA GEMM: C[M,N] = (A @ B + bias) * scale (+ residual)
// A is f16 [M,K] (optionally split at ksplit into two buffers A1/A2 -> fused
// concat), B is f16 [K,N] row-major. 64x64 block tile, 4 waves, K-step 32.
// ---------------------------------------------------------------------------
template <bool OUT_F16>
__launch_bounds__(128)
__global__ void wmma_gemm(const _Float16* __restrict__ A1, int lda1,
                          const _Float16* __restrict__ A2, int lda2, int ksplit,
                          const _Float16* __restrict__ Bm, int ldb,
                          const float* __restrict__ bias,
                          const float* __restrict__ residual,
                          float scale, void* __restrict__ Cout, int ldc, int K) {
  __shared__ _Float16 As[64 * 40];  // 64 rows x 32 K (+8 pad halves)
  __shared__ _Float16 Bt[64 * 40];  // B tile stored transposed: [n][k]

  const int t = threadIdx.x;
  const int tileM = blockIdx.y * 64, tileN = blockIdx.x * 64;
  const int wave = t >> 5, lane = t & 31, hf = lane >> 4, l16 = lane & 15;
  const int mB = (wave >> 1) * 32, nB = (wave & 1) * 32;

  v8f acc[2][2];
  acc[0][0] = zero8(); acc[0][1] = zero8();
  acc[1][0] = zero8(); acc[1][1] = zero8();

  const int tr = t >> 2, tc = (t & 3) * 8;     // A tile loader mapping
  const int bk = t >> 2, bn0 = (t & 3) * 16;   // B tile loader mapping

  for (int k0 = 0; k0 < K; k0 += 32) {
    // Stage A 64x32 (handles fused-concat split; K-tile never straddles split)
    for (int rr = tr; rr < 64; rr += 32) {
      int gk = k0 + tc;
      const _Float16* src = (gk < ksplit)
          ? (A1 + (size_t)(tileM + rr) * lda1 + gk)
          : (A2 + (size_t)(tileM + rr) * lda2 + (gk - ksplit));
      *(v8h*)&As[rr * 40 + tc] = *(const v8h*)src;
    }
    // Stage B 32x64 transposed into Bt[n][k]
    {
      const _Float16* bs = Bm + (size_t)(k0 + bk) * ldb + tileN + bn0;
      v8h b0 = *(const v8h*)bs;
      v8h b1 = *(const v8h*)(bs + 8);
#pragma unroll
      for (int j = 0; j < 8; j++) Bt[(bn0 + j) * 40 + bk] = b0[j];
#pragma unroll
      for (int j = 0; j < 8; j++) Bt[(bn0 + 8 + j) * 40 + bk] = b1[j];
    }
    __syncthreads();

    v16h af[2], bf[2];
#pragma unroll
    for (int mi = 0; mi < 2; mi++) af[mi] = lds_frag(&As[(mB + mi * 16 + l16) * 40], 0, hf);
#pragma unroll
    for (int ni = 0; ni < 2; ni++) bf[ni] = lds_frag(&Bt[(nB + ni * 16 + l16) * 40], 0, hf);
#pragma unroll
    for (int mi = 0; mi < 2; mi++)
#pragma unroll
      for (int ni = 0; ni < 2; ni++)
        acc[mi][ni] = WMMA_F32_F16(af[mi], bf[ni], acc[mi][ni]);
    __syncthreads();
  }

  // Epilogue. C layout: VGPR g -> row = g + 8*half, col = lane%16.
#pragma unroll
  for (int mi = 0; mi < 2; mi++)
#pragma unroll
    for (int ni = 0; ni < 2; ni++)
#pragma unroll
      for (int g = 0; g < 8; g++) {
        int row = tileM + mB + mi * 16 + g + 8 * hf;
        int col = tileN + nB + ni * 16 + l16;
        float v = acc[mi][ni][g];
        if (bias) v += bias[col];
        v *= scale;
        if (residual) v += residual[(size_t)row * ldc + col];
        if (OUT_F16)
          ((_Float16*)Cout)[(size_t)row * ldc + col] = (_Float16)v;
        else
          ((float*)Cout)[(size_t)row * ldc + col] = v;
      }
}

// ---------------------------------------------------------------------------
// Flash attention (one direction): O = softmax(Q K^T) V, per (batch, head).
// Q/K/V/O are f16 in merged [B*N, D] layout, head h owns cols [h*64, h*64+64).
// Block = 64 query rows, 4 waves x 16 rows each; stream keys in tiles of 64.
// K tiles are staged into LDS by the Tensor Data Mover (tensor_load_to_lds,
// TENSORcnt); V tiles are transposed manually; next V tile is prefetched.
// sim (536 MB) is never materialized.
// ---------------------------------------------------------------------------
__launch_bounds__(128)
__global__ void flash_attn(const _Float16* __restrict__ Qm, const _Float16* __restrict__ Km,
                           const _Float16* __restrict__ Vm, _Float16* __restrict__ Om) {
  constexpr int N = 4096, D = 256, Dh = 64;
  const int qTile = blockIdx.x * 64;
  const int h = blockIdx.y, bb = blockIdx.z;
  const _Float16* Qb = Qm + ((size_t)bb * N) * D + h * Dh;
  const _Float16* Kb = Km + ((size_t)bb * N) * D + h * Dh;
  const _Float16* Vb = Vm + ((size_t)bb * N) * D + h * Dh;

  __shared__ _Float16 Qs[64 * 72];  // [query][dh]
  __shared__ _Float16 Ks[64 * 72];  // [key][dh]   (= B^T layout for S = Q K^T)
  __shared__ _Float16 Vt[64 * 72];  // [dh][key]   (= B^T layout for O = P V)
  __shared__ _Float16 Ps[64 * 72];  // P staging, per-wave-private 16-row bands

  const int t = threadIdx.x;
  const int wave = t >> 5, lane = t & 31, hf = lane >> 4, l16 = lane & 15;

  for (int s = t; s < 512; s += 128) {
    int r = s >> 3, c = (s & 7) * 8;
    *(v8h*)&Qs[r * 72 + c] = *(const v8h*)(Qb + (size_t)(qTile + r) * D + c);
  }
  __syncthreads();

  v16h qf[2];
#pragma unroll
  for (int ks = 0; ks < 2; ks++) qf[ks] = lds_frag(&Qs[(wave * 16 + l16) * 72], ks * 32, hf);

  v8f o[4];
#pragma unroll
  for (int ni = 0; ni < 4; ni++) o[ni] = zero8();
  float rowm[8], rowl[8];
#pragma unroll
  for (int g = 0; g < 8; g++) { rowm[g] = -3.0e38f; rowl[g] = 0.f; }

  // TDM descriptor pieces that are loop-invariant (per wave: 16 key rows).
  // D# group1: data_size=2B (code 1), pad_enable=1, pad_interval=32DW (code 4,
  // = 128B = one 64-half tile row), pad_amount=4DW (code 3, = 8 halves) ->
  // reproduces the 72-half LDS row stride. tile = 64 x 16, stride = D = 256.
  const u32 ldsKsBase = (u32)(size_t)(&Ks[0]) + (u32)(wave * 16) * 144u;
  u32x8 tg1;
  tg1[0] = (1u << 16) | (1u << 20) | (4u << 22) | (3u << 25);
  tg1[1] = (64u << 16);            // tensor_dim0 = 64 (bits 79:48, low part)
  tg1[2] = (16u << 16);            // tensor_dim0 hi | tensor_dim1 = 16 rows
  tg1[3] = (64u << 16);            // tensor_dim1 hi | tile_dim0 = 64
  tg1[4] = 16u;                    // tile_dim1 = 16, tile_dim2 = 0
  tg1[5] = (u32)D;                 // tensor_dim0_stride = 256 elements
  tg1[6] = 0u;
  tg1[7] = 0u;

  for (int kt0 = 0; kt0 < N; kt0 += 64) {
    __syncthreads();  // previous tile fully consumed by all waves

    // Issue TDM: this wave's 16 rows of the K tile -> Ks (async, TENSORcnt)
    {
      u64 ga = (u64)(size_t)(Kb + (size_t)(kt0 + wave * 16) * D);
      u32x4 tg0;
      tg0[0] = 1u;                              // count=1, user mode
      tg0[1] = ldsKsBase;                       // lds_addr
      tg0[2] = (u32)ga;                         // global_addr[31:0]
      tg0[3] = (u32)(ga >> 32) | (2u << 30);    // global_addr[56:32] | type=2
      asm volatile("tensor_load_to_lds %0, %1" :: "s"(tg0), "s"(tg1) : "memory");
    }

    // Manual transpose staging of the V tile (overlaps the TDM transfer)
    for (int s = t; s < 512; s += 128) {
      int r = s >> 3, c = (s & 7) * 8;
      v8h vv = *(const v8h*)(Vb + (size_t)(kt0 + r) * D + c);
#pragma unroll
      for (int j = 0; j < 8; j++) Vt[(c + j) * 72 + r] = vv[j];  // transpose
    }
    // Warm GL2 with the next V tile (one row per pair of lanes)
    if (kt0 + 64 < N)
      __builtin_prefetch(Vb + (size_t)(kt0 + 64 + (t >> 1)) * D, 0, 1);

    __builtin_amdgcn_s_wait_tensorcnt(0);  // K tile landed in LDS
    __syncthreads();

    // S tile: 16 queries x 64 keys per wave (Dh=64 -> 2 WMMA K-steps)
    v8f sf[4];
#pragma unroll
    for (int ni = 0; ni < 4; ni++) {
      sf[ni] = zero8();
#pragma unroll
      for (int ks = 0; ks < 2; ks++) {
        v16h kf = lds_frag(&Ks[(ni * 16 + l16) * 72], ks * 32, hf);
        sf[ni] = WMMA_F32_F16(qf[ks], kf, sf[ni]);
      }
    }

    // Online softmax. Row m = g + 8*hf lives in VGPR g across lanes of this
    // half-group; reduce across the 16-lane half with shfl_xor.
    float alphaV[8];
#pragma unroll
    for (int g = 0; g < 8; g++) {
      float mx = fmaxf(fmaxf(sf[0][g], sf[1][g]), fmaxf(sf[2][g], sf[3][g]));
#pragma unroll
      for (int off = 1; off < 16; off <<= 1) mx = fmaxf(mx, __shfl_xor(mx, off, 32));
      float mnew = fmaxf(rowm[g], mx);
      float al = __expf(rowm[g] - mnew);
      float ps = 0.f;
#pragma unroll
      for (int ni = 0; ni < 4; ni++) {
        float p = __expf(sf[ni][g] - mnew);
        sf[ni][g] = p;
        ps += p;
      }
#pragma unroll
      for (int off = 1; off < 16; off <<= 1) ps += __shfl_xor(ps, off, 32);
      rowl[g] = rowl[g] * al + ps;
      rowm[g] = mnew;
      alphaV[g] = al;
    }

    // C-layout -> A-layout for P via per-wave LDS band (same-wave DS in order)
#pragma unroll
    for (int ni = 0; ni < 4; ni++)
#pragma unroll
      for (int g = 0; g < 8; g++)
        Ps[(wave * 16 + g + 8 * hf) * 72 + ni * 16 + l16] = (_Float16)sf[ni][g];

#pragma unroll
    for (int ni = 0; ni < 4; ni++)
#pragma unroll
      for (int g = 0; g < 8; g++) o[ni][g] *= alphaV[g];

    v16h pf[2];
#pragma unroll
    for (int ks = 0; ks < 2; ks++) pf[ks] = lds_frag(&Ps[(wave * 16 + l16) * 72], ks * 32, hf);
#pragma unroll
    for (int ni = 0; ni < 4; ni++)
#pragma unroll
      for (int ks = 0; ks < 2; ks++) {
        v16h vf = lds_frag(&Vt[(ni * 16 + l16) * 72], ks * 32, hf);
        o[ni] = WMMA_F32_F16(pf[ks], vf, o[ni]);
      }
  }

#pragma unroll
  for (int ni = 0; ni < 4; ni++)
#pragma unroll
    for (int g = 0; g < 8; g++) {
      int row = qTile + wave * 16 + g + 8 * hf;
      int col = h * Dh + ni * 16 + l16;
      Om[((size_t)bb * N + row) * D + col] = (_Float16)(o[ni][g] / rowl[g]);
    }
}

// ---------------------------------------------------------------------------
// Elementwise helpers
// ---------------------------------------------------------------------------
__global__ void cvt_f32_f16(const float* __restrict__ in, _Float16* __restrict__ out, int n) {
  int i = blockIdx.x * blockDim.x + threadIdx.x;
  if (i < n) out[i] = (_Float16)in[i];
}

// LayerNorm(512) + exact GELU, one block per row, f32 in -> f16 out
__launch_bounds__(256)
__global__ void ln_gelu_kernel(const float* __restrict__ H, const float* __restrict__ gamma,
                               const float* __restrict__ beta, _Float16* __restrict__ G) {
  const int row = blockIdx.x;
  const int t = threadIdx.x;
  const float* hr = H + (size_t)row * 512;
  float a = hr[t], b = hr[t + 256];
  float s = a + b, s2 = a * a + b * b;
#pragma unroll
  for (int off = 16; off > 0; off >>= 1) {
    s += __shfl_xor(s, off, 32);
    s2 += __shfl_xor(s2, off, 32);
  }
  __shared__ float w1[8], w2[8];
  int w = t >> 5, ln = t & 31;
  if (ln == 0) { w1[w] = s; w2[w] = s2; }
  __syncthreads();
  if (w == 0) {
    float u = (ln < 8) ? w1[ln] : 0.f;
    float u2 = (ln < 8) ? w2[ln] : 0.f;
#pragma unroll
    for (int off = 4; off > 0; off >>= 1) {
      u += __shfl_xor(u, off, 32);
      u2 += __shfl_xor(u2, off, 32);
    }
    if (ln == 0) { w1[0] = u; w2[0] = u2; }
  }
  __syncthreads();
  float mean = w1[0] * (1.f / 512.f);
  float var = w2[0] * (1.f / 512.f) - mean * mean;
  float rs = rsqrtf(var + 1e-5f);
#pragma unroll
  for (int e = 0; e < 2; e++) {
    int c = t + e * 256;
    float x = (e == 0 ? a : b);
    float hn = (x - mean) * rs * gamma[c] + beta[c];
    float ge = 0.5f * hn * (1.f + erff(hn * 0.70710678118654752f));
    G[(size_t)row * 512 + c] = (_Float16)ge;
  }
}

// ---------------------------------------------------------------------------
// Host orchestration
// ---------------------------------------------------------------------------
extern "C" void kernel_launch(void* const* d_in, const int* in_sizes, int n_in,
                              void* d_out, int out_size, void* d_ws, size_t ws_size,
                              hipStream_t stream) {
  (void)in_sizes; (void)n_in; (void)out_size; (void)ws_size;
  const float* x0   = (const float*)d_in[0];
  const float* x1   = (const float*)d_in[1];
  const float* Wqk  = (const float*)d_in[2];
  const float* bqk  = (const float*)d_in[3];
  const float* Wv   = (const float*)d_in[4];
  const float* bv   = (const float*)d_in[5];
  const float* Wo   = (const float*)d_in[6];
  const float* bo   = (const float*)d_in[7];
  const float* W1   = (const float*)d_in[8];
  const float* b1   = (const float*)d_in[9];
  const float* ln_g = (const float*)d_in[10];
  const float* ln_b = (const float*)d_in[11];
  const float* W2   = (const float*)d_in[12];
  const float* b2   = (const float*)d_in[13];

  constexpr int Bn = 2, N = 4096, D = 256, D2 = 512, H = 4;
  constexpr int M = Bn * N;  // 8192 tokens
  const float s_half = 0.35355339059327379f;  // (Dh^-0.5)^0.5, Dh=64

  char* ws = (char*)d_ws;
  size_t off = 0;
  auto alloc = [&](size_t bytes) -> void* {
    void* p = ws + off;
    off += (bytes + 255) & ~(size_t)255;
    return p;
  };

  _Float16* Xh0  = (_Float16*)alloc((size_t)M * D * 2);
  _Float16* Xh1  = (_Float16*)alloc((size_t)M * D * 2);
  _Float16* WqkH = (_Float16*)alloc((size_t)D * D * 2);
  _Float16* WvH  = (_Float16*)alloc((size_t)D * D * 2);
  _Float16* WoH  = (_Float16*)alloc((size_t)D * D * 2);
  _Float16* W1H  = (_Float16*)alloc((size_t)D2 * D2 * 2);
  _Float16* W2H  = (_Float16*)alloc((size_t)D2 * D * 2);
  _Float16* qk0  = (_Float16*)alloc((size_t)M * D * 2);
  _Float16* qk1  = (_Float16*)alloc((size_t)M * D * 2);
  _Float16* v0h  = (_Float16*)alloc((size_t)M * D * 2);
  _Float16* v1h  = (_Float16*)alloc((size_t)M * D * 2);
  _Float16* m0   = (_Float16*)alloc((size_t)M * D * 2);
  _Float16* m1   = (_Float16*)alloc((size_t)M * D * 2);
  _Float16* p0   = (_Float16*)alloc((size_t)M * D * 2);
  _Float16* p1   = (_Float16*)alloc((size_t)M * D * 2);
  float*    hbuf = (float*)alloc((size_t)M * D2 * 4);
  _Float16* gbuf = (_Float16*)alloc((size_t)M * D2 * 2);

  auto cvt = [&](const float* src, _Float16* dst, int n) {
    cvt_f32_f16<<<dim3((n + 255) / 256), dim3(256), 0, stream>>>(src, dst, n);
  };
  cvt(x0, Xh0, M * D);
  cvt(x1, Xh1, M * D);
  cvt(Wqk, WqkH, D * D);
  cvt(Wv, WvH, D * D);
  cvt(Wo, WoH, D * D);
  cvt(W1, W1H, D2 * D2);
  cvt(W2, W2H, D2 * D);

  const dim3 blk(128);
  const dim3 gP(D / 64, M / 64);    // [8192,256]x[256,256]
  const dim3 gF(D2 / 64, M / 64);   // [8192,512]x[512,512]

  // QK / V projections (qk pre-scaled by s on both sides -> product scaled 1/8)
  wmma_gemm<true><<<gP, blk, 0, stream>>>(Xh0, D, nullptr, 0, D, WqkH, D, bqk, nullptr, s_half, qk0, D, D);
  wmma_gemm<true><<<gP, blk, 0, stream>>>(Xh1, D, nullptr, 0, D, WqkH, D, bqk, nullptr, s_half, qk1, D, D);
  wmma_gemm<true><<<gP, blk, 0, stream>>>(Xh0, D, nullptr, 0, D, WvH, D, bv, nullptr, 1.f, v0h, D, D);
  wmma_gemm<true><<<gP, blk, 0, stream>>>(Xh1, D, nullptr, 0, D, WvH, D, bv, nullptr, 1.f, v1h, D, D);

  // Both attention directions as streaming flash passes (sim never stored)
  flash_attn<<<dim3(N / 64, H, Bn), blk, 0, stream>>>(qk0, qk1, v1h, m0);  // softmax over N1
  flash_attn<<<dim3(N / 64, H, Bn), blk, 0, stream>>>(qk1, qk0, v0h, m1);  // softmax over N0

  // Output projection
  wmma_gemm<true><<<gP, blk, 0, stream>>>(m0, D, nullptr, 0, D, WoH, D, bo, nullptr, 1.f, p0, D, D);
  wmma_gemm<true><<<gP, blk, 0, stream>>>(m1, D, nullptr, 0, D, WoH, D, bo, nullptr, 1.f, p1, D, D);

  float* y0 = (float*)d_out;
  float* y1 = y0 + (size_t)M * D;

  // FFN tensor 0: concat fused via split-A GEMM; LN+GELU; W2 + residual
  wmma_gemm<false><<<gF, blk, 0, stream>>>(Xh0, D, p0, D, D, W1H, D2, b1, nullptr, 1.f, hbuf, D2, D2);
  ln_gelu_kernel<<<dim3(M), dim3(256), 0, stream>>>(hbuf, ln_g, ln_b, gbuf);
  wmma_gemm<false><<<gP, blk, 0, stream>>>(gbuf, D2, nullptr, 0, D2, W2H, D, b2, x0, 1.f, y0, D, D2);

  // FFN tensor 1 (reuses hbuf/gbuf; stream order serializes)
  wmma_gemm<false><<<gF, blk, 0, stream>>>(Xh1, D, p1, D, D, W1H, D2, b1, nullptr, 1.f, hbuf, D2, D2);
  ln_gelu_kernel<<<dim3(M), dim3(256), 0, stream>>>(hbuf, ln_g, ln_b, gbuf);
  wmma_gemm<false><<<gP, blk, 0, stream>>>(gbuf, D2, nullptr, 0, D2, W2H, D, b2, x1, 1.f, y1, D, D2);
}
